// QMogrifierCell_6511170421079
// MI455X (gfx1250) — matrix operand order repaired and checked by reading, not compiled, not run on hardware
//
#include <hip/hip_runtime.h>

typedef unsigned short u16;
typedef __attribute__((ext_vector_type(16))) __bf16 v16bf;
typedef __attribute__((ext_vector_type(8)))  float  v8f;
typedef __attribute__((ext_vector_type(4)))  unsigned int v4u;

union BfTile { v16bf v; v4u q[2]; };

static __device__ __forceinline__ u16 f2bf(float f) {
    unsigned u = __float_as_uint(f);
    unsigned r = u + 0x7FFFu + ((u >> 16) & 1u);   // round-to-nearest-even
    return (u16)(r >> 16);
}
static __device__ __forceinline__ float sigm(float x) {
    return 1.0f / (1.0f + __expf(-x));
}

// ---- CDNA5 async Global -> LDS copies (ASYNCcnt-tracked, no VGPR round trip)
static __device__ __forceinline__ void async_g2l_b128(void* lds, const void* g) {
    unsigned l = (unsigned)(size_t)lds;              // generic LDS addr[31:0] = LDS offset
    asm volatile("global_load_async_to_lds_b128 %0, %1, off" :: "v"(l), "v"(g) : "memory");
}
static __device__ __forceinline__ void async_g2l_b64(void* lds, const void* g) {
    unsigned l = (unsigned)(size_t)lds;
    asm volatile("global_load_async_to_lds_b64 %0, %1, off" :: "v"(l), "v"(g) : "memory");
}
static __device__ __forceinline__ void wait_async0() {
    asm volatile("s_wait_asynccnt 0" ::: "memory");
}

// A tile: 16x32 bf16 from row-major buffer, row stride 1024 elements.
// ISA layout: lane L (m = L&15): VGPR0..3 hold K = kb + (L>>4)*8 + {0..7},
//             VGPR4..7 hold K = kb + 16 + (L>>4)*8 + {0..7}.
static __device__ __forceinline__ v16bf load_a_tile(const u16* __restrict__ A,
                                                    int row, int kb, int lane) {
    const u16* p = A + (size_t)row * 1024 + kb + ((lane >> 4) << 3);
    BfTile t;
    t.q[0] = *(const v4u*)p;
    t.q[1] = *(const v4u*)(p + 16);
    return t.v;
}

// B tile from LDS: row-major [16][40] tile holding 16 rows x 32 k (pitch 40
// elements = 80B -> conflict-free bank stride for the 16-lane access pattern).
// Lane L (n = L&15) needs K = (L>>4)*8 + {0..7}, then 16 + (L>>4)*8 + {0..7}.
static __device__ __forceinline__ v16bf load_b_lds(const u16* p16, int lane) {
    const u16* p = p16 + (size_t)(lane & 15) * 40 + ((lane >> 4) << 3);
    BfTile t;
    t.q[0] = *(const v4u*)p;
    t.q[1] = *(const v4u*)(p + 16);
    return t.v;
}

#define WMMA_BF16(A, B, C) \
    __builtin_amdgcn_wmma_f32_16x16x32_bf16(false, (A), false, (B), (short)0, (C), false, false)

// ---------------- one-time setup kernels ----------------

// Wc[n][k] (bf16, stride 2048) = bf16( k<1024 ? W_ih[n][k] : W_hh[n][k-1024] )
__global__ void build_wc(const float* __restrict__ Wih, const float* __restrict__ Whh,
                         u16* __restrict__ Wc) {
    int idx = blockIdx.x * 256 + threadIdx.x;        // 4096*2048 elements
    int n = idx >> 11, k = idx & 2047;
    float v = (k < 1024) ? Wih[n * 1024 + k] : Whh[n * 1024 + (k - 1024)];
    Wc[idx] = f2bf(v);
}

// Wm[i][k] (bf16, stride 1024) = bf16(W_mod[i][k]) for k<1024 (tail cols in fp32)
__global__ void build_wmod(const float* __restrict__ Wmod, u16* __restrict__ Wm) {
    int idx = blockIdx.x * 256 + threadIdx.x;        // 1024*1024 elements
    int i = idx >> 10, k = idx & 1023;
    Wm[idx] = f2bf(Wmod[i * 1026 + k]);
}

// zero c and h(parity 0), fold biases
__global__ void init_state(const float* __restrict__ bih, const float* __restrict__ bhh,
                           float* __restrict__ bias, float* __restrict__ c,
                           u16* __restrict__ h0) {
    int idx = blockIdx.x * 256 + threadIdx.x;        // 65536 threads
    c[idx] = 0.0f;
    h0[idx] = (u16)0;
    if (idx < 4096) bias[idx] = bih[idx] + bhh[idx];
}

__global__ void final_copy(const float* __restrict__ hs, float* __restrict__ hfin) {
    int idx = blockIdx.x * 256 + threadIdx.x;        // 65536 threads
    int b = idx >> 10, j = idx & 1023;
    hfin[idx] = hs[((size_t)b * 512 + 511) * 1024 + j];
}

// ---------------- per-timestep kernels ----------------

// Kernel A: m = sigmoid(h @ Wm^T + d*Wmod[:,1024] + lq*Wmod[:,1025] + b_mod);
//           mx[b][i] = bf16(m * seq[b][t][i]).
// grid = 64 blocks (one jblk each), block = 128 (wave w = M-tile w).
// Wm tile (16x32 bf16) for the block's 16 output columns is staged once per
// K-step into LDS via async global->LDS (double buffered), shared by 4 waves.
__global__ __launch_bounds__(128)
void mod_gemm(const float* __restrict__ seq, const float* __restrict__ dsq,
              const float* __restrict__ lqs, const float* __restrict__ Wmod,
              const float* __restrict__ bmod, const u16* __restrict__ Wm,
              const u16* __restrict__ h_in, u16* __restrict__ mx, int t) {
    __shared__ __align__(16) u16 Bsh[2][16][40];

    const int tid   = threadIdx.x;
    const int lane  = tid & 31;
    const int mtile = tid >> 5;
    const int jblk  = blockIdx.x;

    // staging map: thread moves 8B (4 bf16): row = tid>>3, part = (tid&7)*4 elems
    const int srow  = tid >> 3;
    const int spart = (tid & 7) << 2;
    const u16* sgl  = Wm + (size_t)(jblk * 16 + srow) * 1024 + spart;
    u16* sl0        = &Bsh[0][srow][spart];
    u16* sl1        = &Bsh[1][srow][spart];

    v8f acc = {0.f, 0.f, 0.f, 0.f, 0.f, 0.f, 0.f, 0.f};
    const int arow = mtile * 16 + (lane & 15);

    async_g2l_b64(sl0, sgl);                         // stage k-step 0
    wait_async0();
    __syncthreads();

    for (int kbi = 0; kbi < 32; ++kbi) {
        const int cur = kbi & 1;
        if (kbi + 1 < 32)
            async_g2l_b64(cur ? sl0 : sl1, sgl + (kbi + 1) * 32);
        v16bf a = load_a_tile(h_in, arow, kbi * 32, lane);
        v16bf b = load_b_lds(&Bsh[cur][0][0], lane);
        acc = WMMA_BF16(a, b, acc);
        wait_async0();
        __syncthreads();
    }

    const int i     = jblk * 16 + (lane & 15);
    const float wc0 = Wmod[(size_t)i * 1026 + 1024];
    const float wc1 = Wmod[(size_t)i * 1026 + 1025];
    const float bm  = bmod[i];
    const int mbase = mtile * 16 + ((lane >> 4) << 3);
#pragma unroll
    for (int r = 0; r < 8; ++r) {
        int brow = mbase + r;
        float d  = dsq[brow * 512 + t];
        float lq = lqs[brow * 512 + t];
        float mv = sigm(acc[r] + d * wc0 + lq * wc1 + bm);
        float x  = seq[((size_t)brow * 512 + t) * 1024 + i];
        mx[brow * 1024 + i] = f2bf(mv * x);
    }
}

// Kernel B: gates = [mx|h] @ Wc^T + bias; LSTM elementwise; write c, hs, h_bf16.
// grid = 64 blocks (one jblk each), block = 128 (wave w = M-tile w).
// Per K-step the 4 gate B-tiles (4KB) are staged once into LDS via async
// global->LDS (double buffered) and shared by all 4 waves.
__global__ __launch_bounds__(128)
void gates_lstm(const u16* __restrict__ Wc, const float* __restrict__ bias,
                const u16* __restrict__ mx, const u16* __restrict__ h_in,
                u16* __restrict__ h_out, float* __restrict__ c,
                float* __restrict__ hs, int t) {
    __shared__ __align__(16) u16 Bsh[2][4][16][40];

    const int tid   = threadIdx.x;
    const int lane  = tid & 31;
    const int mtile = tid >> 5;
    const int jblk  = blockIdx.x;

    // staging map: thread moves 32B (16 bf16): gate = tid>>5, row = (tid>>1)&15,
    // half = (tid&1)*16 elems. Two async b128 per thread.
    const int sg    = tid >> 5;
    const int srow  = (tid >> 1) & 15;
    const int shalf = (tid & 1) << 4;
    const u16* sgl  = Wc + (size_t)(sg * 1024 + jblk * 16 + srow) * 2048 + shalf;
    u16* sl0        = &Bsh[0][sg][srow][shalf];
    u16* sl1        = &Bsh[1][sg][srow][shalf];

    v8f a0 = {0.f,0.f,0.f,0.f,0.f,0.f,0.f,0.f};
    v8f a1 = a0, a2 = a0, a3 = a0;
    const int arow = mtile * 16 + (lane & 15);
    const int j    = jblk * 16 + (lane & 15);

    async_g2l_b128(sl0, sgl);                        // stage k-step 0 (full 32B)
    async_g2l_b128(sl0 + 8, sgl + 8);
    wait_async0();
    __syncthreads();

    for (int kbi = 0; kbi < 64; ++kbi) {
        const int cur = kbi & 1;
        if (kbi + 1 < 64) {
            const u16* g = sgl + (kbi + 1) * 32;
            u16* l = cur ? sl0 : sl1;
            async_g2l_b128(l, g);
            async_g2l_b128(l + 8, g + 8);
        }
        const int kb = kbi * 32;
        // A operand: first 1024 K from m*x, next 1024 K from h
        const u16* Abuf = (kb < 1024) ? mx : h_in;
        const int  koff = (kb < 1024) ? kb : (kb - 1024);
        v16bf a  = load_a_tile(Abuf, arow, koff, lane);
        v16bf b0 = load_b_lds(&Bsh[cur][0][0][0], lane);
        v16bf b1 = load_b_lds(&Bsh[cur][1][0][0], lane);
        v16bf b2 = load_b_lds(&Bsh[cur][2][0][0], lane);
        v16bf b3 = load_b_lds(&Bsh[cur][3][0][0], lane);
        a0 = WMMA_BF16(a, b0, a0);
        a1 = WMMA_BF16(a, b1, a1);
        a2 = WMMA_BF16(a, b2, a2);
        a3 = WMMA_BF16(a, b3, a3);
        wait_async0();
        __syncthreads();
    }

    const float bi = bias[j];
    const float bf = bias[1024 + j];
    const float bg = bias[2048 + j];
    const float bo = bias[3072 + j];
    const int mbase = mtile * 16 + ((lane >> 4) << 3);
#pragma unroll
    for (int r = 0; r < 8; ++r) {
        int brow = mbase + r;
        float gi = a0[r] + bi;
        float gf = a1[r] + bf;
        float gg = a2[r] + bg;
        float go = a3[r] + bo;
        float cc = c[brow * 1024 + j];
        float cn = sigm(gf) * cc + sigm(gi) * tanhf(gg);
        float hn = sigm(go) * tanhf(cn);
        c[brow * 1024 + j] = cn;
        hs[((size_t)brow * 512 + t) * 1024 + j] = hn;
        h_out[brow * 1024 + j] = f2bf(hn);
    }
}

extern "C" void kernel_launch(void* const* d_in, const int* in_sizes, int n_in,
                              void* d_out, int out_size, void* d_ws, size_t ws_size,
                              hipStream_t stream) {
    const float* seq  = (const float*)d_in[0];   // [64,512,1024]
    const float* dsq  = (const float*)d_in[1];   // [64,512,1]
    const float* lqs  = (const float*)d_in[2];   // [64,512,1]
    const float* Wih  = (const float*)d_in[3];   // [4096,1024]
    const float* Whh  = (const float*)d_in[4];   // [4096,1024]
    const float* bih  = (const float*)d_in[5];   // [4096]
    const float* bhh  = (const float*)d_in[6];   // [4096]
    const float* Wmod = (const float*)d_in[7];   // [1024,1026]
    const float* bmod = (const float*)d_in[8];   // [1024]

    float* out  = (float*)d_out;                 // hs [64,512,1024] ++ h [64,1024]
    float* hfin = out + (size_t)64 * 512 * 1024;

    char* ws = (char*)d_ws;
    size_t off = 0;
    u16*   Wc   = (u16*)(ws + off);   off += (size_t)4096 * 2048 * 2;  // 16 MB bf16 [Wih|Whh]
    u16*   Wm   = (u16*)(ws + off);   off += (size_t)1024 * 1024 * 2;  // 2 MB bf16 W_mod[:, :1024]
    float* bias = (float*)(ws + off); off += (size_t)4096 * 4;
    u16*   mx   = (u16*)(ws + off);   off += (size_t)65536 * 2;        // m*x bf16
    u16*   hb0  = (u16*)(ws + off);   off += (size_t)65536 * 2;        // h bf16, parity 0
    u16*   hb1  = (u16*)(ws + off);   off += (size_t)65536 * 2;        // h bf16, parity 1
    float* cbuf = (float*)(ws + off); off += (size_t)65536 * 4;        // c state fp32

    build_wc  <<<(4096 * 2048) / 256, 256, 0, stream>>>(Wih, Whh, Wc);
    build_wmod<<<(1024 * 1024) / 256, 256, 0, stream>>>(Wmod, Wm);
    init_state<<<65536 / 256, 256, 0, stream>>>(bih, bhh, bias, cbuf, hb0);

    for (int t = 0; t < 512; ++t) {
        const u16* hin = (t & 1) ? hb1 : hb0;
        u16*       hot = (t & 1) ? hb0 : hb1;
        mod_gemm <<<64, 128, 0, stream>>>(seq, dsq, lqs, Wmod, bmod, Wm, hin, mx, t);
        gates_lstm<<<64, 128, 0, stream>>>(Wc, bias, mx, hin, hot, cbuf, out, t);
    }

    final_copy<<<65536 / 256, 256, 0, stream>>>(out, hfin);
}
